// InnerShiftSingle_13030930776852
// MI455X (gfx1250) — matrix operand order repaired and checked by reading, not compiled, not run on hardware
//
#include <hip/hip_runtime.h>
#include <stdint.h>

#define BZ   8
#define CH   512
#define C2   256
#define HW   4096      // 64*64
#define QS   8         // q-split across blockIdx.y
#define PBLK 128       // p rows per workgroup (8 waves x 16)
#define NT   256
#define BSTRIDE 264    // padded LDS row stride (ushorts) to avoid bank conflicts

typedef __bf16 bf16;
typedef __attribute__((ext_vector_type(16))) bf16  v16bf;
typedef __attribute__((ext_vector_type(8)))  bf16  v8bf;
typedef __attribute__((ext_vector_type(8)))  float v8f;

// ---------------- workspace init ----------------
__global__ void init_ws_kernel(unsigned long long* __restrict__ keys,
                               int* __restrict__ counters) {
  int i = blockIdx.x * blockDim.x + threadIdx.x;
  if (i < BZ * HW) keys[i] = 0ULL;
  if (i < 2) counters[i] = 0;
}

// ---------------- compaction of flagged p and valid q-tiles ----------------
__global__ void compact_kernel(const int* __restrict__ mask,
                               int* __restrict__ pidx,
                               int* __restrict__ qbase,
                               int* __restrict__ counters) {
  int t = blockIdx.x * blockDim.x + threadIdx.x;
  if (t < HW) {
    if (mask[t] == 1) {
      int s = atomicAdd(&counters[0], 1);
      pidx[s] = t;
    }
  }
  if (t < HW / 16) {
    bool any_valid = false;
    for (int j = 0; j < 16; ++j) any_valid |= (mask[t * 16 + j] == 0);
    if (any_valid) {
      int s = atomicAdd(&counters[1], 1);
      qbase[s] = t * 16;
    }
  }
}

// ---------------- inverse norms of latter columns ----------------
__global__ void norm_kernel(const float* __restrict__ x,
                            float* __restrict__ invn) {
  int i = blockIdx.x * blockDim.x + threadIdx.x;  // b*HW + q
  if (i >= BZ * HW) return;
  int b = i >> 12, q = i & (HW - 1);
  const float* p = x + (size_t)b * CH * HW + (size_t)C2 * HW + q;
  float s = 0.0f;
  for (int c = 0; c < C2; ++c) {
    float v = p[(size_t)c * HW];
    s += v * v;
  }
  invn[i] = 1.0f / (sqrtf(s) + 1e-8f);
}

// ---------------- fused WMMA GEMM + masked argmax ----------------
__global__ __launch_bounds__(NT)
void gemm_argmax_kernel(const float* __restrict__ x,
                        const int* __restrict__ mask,
                        const float* __restrict__ invn,
                        const int* __restrict__ pidx,
                        const int* __restrict__ qbase,
                        const int* __restrict__ counters,
                        unsigned long long* __restrict__ keys) {
  const int npos = counters[0];
  const int nqt  = counters[1];
  const int b    = blockIdx.x >> 5;        // 32 p-blocks per batch
  const int pblk = blockIdx.x & 31;
  const int pstart = pblk * PBLK;
  if (pstart >= npos || nqt == 0) return;

  const int tid  = threadIdx.x;
  const int lane = tid & 31;
  const int wave = tid >> 5;
  const int g    = lane >> 4;              // half-wave group
  const int n16  = lane & 15;

  const float* xb_former = x + (size_t)b * CH * HW;       // channels [0,256)
  const float* xb_latter = xb_former + (size_t)C2 * HW;   // channels [256,512)

  // ---- A fragments for this wave's 16 p-rows, bf16 hi/lo split, K=256 ----
  int prow = pstart + wave * 16 + n16;
  if (prow >= npos) prow = npos - 1;       // duplicate rows are idempotent
  const int pp = pidx[prow];

  v16bf ahi[8], alo[8];
#pragma unroll
  for (int ks = 0; ks < 8; ++ks) {
#pragma unroll
    for (int e = 0; e < 16; ++e) {
      int kk = e + 16 * g;                            // contiguous k order (matches B)
      int Kc = ks * 32 + kk;
      float f = xb_former[(size_t)Kc * HW + pp];
      unsigned int u = __float_as_uint(f);
      float hf = __uint_as_float(u & 0xFFFF0000u);
      float lf = f - hf;
      ahi[ks][e] = __builtin_bit_cast(bf16, (unsigned short)(u >> 16));
      alo[ks][e] = __builtin_bit_cast(bf16, (unsigned short)(__float_as_uint(lf) >> 16));
    }
  }

  // double-buffered B tile (hi/lo), [q][K] with padded stride
  __shared__ __attribute__((aligned(32))) unsigned short blds_hi[2][16 * BSTRIDE];
  __shared__ __attribute__((aligned(32))) unsigned short blds_lo[2][16 * BSTRIDE];

  // B staging ownership: each thread owns K-row pair (K2, K2+1), one q-half
  const int tq = tid & 1;
  const int Kp = tid >> 1;
  const int K2 = Kp * 2;
  const int qb = tq * 8;

  float bestv[8];
  int   bestq[8];
#pragma unroll
  for (int r = 0; r < 8; ++r) { bestv[r] = -__builtin_inff(); bestq[r] = 0; }

  float4 r0, r1, r2, r3;                   // prefetch registers (2 rows x 8 q)
  int it = blockIdx.y;
  int Qcur = 0;
  if (it < nqt) {
    Qcur = qbase[it];
    const float* s0 = xb_latter + (size_t)K2 * HW + Qcur + qb;
    r0 = *(const float4*)(s0);
    r1 = *(const float4*)(s0 + 4);
    r2 = *(const float4*)(s0 + HW);
    r3 = *(const float4*)(s0 + HW + 4);
  }

  int buf = 0;
  while (it < nqt) {
    // ---- store prefetched tile to LDS (paired b32 stores) ----
    {
      unsigned int* hi32 = (unsigned int*)(blds_hi[buf]);
      unsigned int* lo32 = (unsigned int*)(blds_lo[buf]);
      float a0[8] = {r0.x, r0.y, r0.z, r0.w, r1.x, r1.y, r1.z, r1.w};
      float a1[8] = {r2.x, r2.y, r2.z, r2.w, r3.x, r3.y, r3.z, r3.w};
#pragma unroll
      for (int j = 0; j < 8; ++j) {
        int q = qb + j;
        unsigned int u0 = __float_as_uint(a0[j]);
        unsigned int u1 = __float_as_uint(a1[j]);
        float l0 = a0[j] - __uint_as_float(u0 & 0xFFFF0000u);
        float l1 = a1[j] - __uint_as_float(u1 & 0xFFFF0000u);
        int idx = (q * BSTRIDE + K2) >> 1;
        hi32[idx] = (u0 >> 16) | (u1 & 0xFFFF0000u);
        lo32[idx] = (__float_as_uint(l0) >> 16) | (__float_as_uint(l1) & 0xFFFF0000u);
      }
    }
    __syncthreads();

    // ---- prefetch next tile into registers (latency hidden by WMMAs) ----
    const int Qthis = Qcur;
    const int itn = it + QS;
    if (itn < nqt) {
      Qcur = qbase[itn];
      const float* s0 = xb_latter + (size_t)K2 * HW + Qcur + qb;
      r0 = *(const float4*)(s0);
      r1 = *(const float4*)(s0 + 4);
      r2 = *(const float4*)(s0 + HW);
      r3 = *(const float4*)(s0 + HW + 4);
    }

    // ---- 16x16 tile: 8 K-steps x 3 WMMA (bf16 split emulation of f32) ----
    v8f acc = {};
#pragma unroll
    for (int ks = 0; ks < 8; ++ks) {
      const int off = n16 * BSTRIDE + ks * 32 + g * 16;   // 16B-aligned
      const unsigned short* bh = &blds_hi[buf][off];
      const unsigned short* bl = &blds_lo[buf][off];
      v8bf bh0 = *(const v8bf*)(bh);
      v8bf bh1 = *(const v8bf*)(bh + 8);
      v8bf bl0 = *(const v8bf*)(bl);
      v8bf bl1 = *(const v8bf*)(bl + 8);
      v16bf bhi = __builtin_shufflevector(bh0, bh1, 0,1,2,3,4,5,6,7,8,9,10,11,12,13,14,15);
      v16bf blo = __builtin_shufflevector(bl0, bl1, 0,1,2,3,4,5,6,7,8,9,10,11,12,13,14,15);
      acc = __builtin_amdgcn_wmma_f32_16x16x32_bf16(false, ahi[ks], false, bhi,
                                                    (short)0, acc, false, false);
      acc = __builtin_amdgcn_wmma_f32_16x16x32_bf16(false, ahi[ks], false, blo,
                                                    (short)0, acc, false, false);
      acc = __builtin_amdgcn_wmma_f32_16x16x32_bf16(false, alo[ks], false, bhi,
                                                    (short)0, acc, false, false);
    }

    // ---- masked running argmax (per-lane column q = Qthis + n16) ----
    const int q = Qthis + n16;
    const float s = invn[b * HW + q];
    if (mask[q] == 0) {
#pragma unroll
      for (int r = 0; r < 8; ++r) {
        float v = acc[r] * s;
        if (v > bestv[r] || (v == bestv[r] && q < bestq[r])) {
          bestv[r] = v; bestq[r] = q;
        }
      }
    }

    it = itn;
    buf ^= 1;
  }

  // ---- argmax reduce across the 16 lanes of each C/D column group ----
#pragma unroll
  for (int r = 0; r < 8; ++r) {
    float v = bestv[r];
    int   q = bestq[r];
    for (int off = 1; off < 16; off <<= 1) {
      float ov = __shfl_xor(v, off, 32);
      int   oq = __shfl_xor(q, off, 32);
      if (ov > v || (ov == v && oq < q)) { v = ov; q = oq; }
    }
    if (n16 == 0) {
      int M  = g * 8 + r;                       // row within this wave's tile
      int pr = pstart + wave * 16 + M;
      if (pr < npos) {
        int pg = pidx[pr];
        unsigned int u = __float_as_uint(v);
        unsigned int mono = (u & 0x80000000u) ? ~u : (u | 0x80000000u);
        unsigned long long key =
            ((unsigned long long)mono << 32) |
            (unsigned long long)(0xFFFFFFFFu - (unsigned int)q);
        atomicMax(&keys[(size_t)b * HW + pg], key);
      }
    }
  }
}

// ---------------- copy former+latter into output ----------------
__global__ void copy_kernel(const float* __restrict__ x, float* __restrict__ out) {
  int i = blockIdx.x * blockDim.x + threadIdx.x;   // float4 index
  const int n = BZ * CH * HW / 4;
  if (i >= n) return;
  const int per_b = CH * HW / 4;
  int b = i / per_b, r = i % per_b;
  const float4* src = (const float4*)x + (size_t)b * per_b + r;
  float4* dst = (float4*)out + (size_t)b * ((CH + C2) * HW / 4) + r;
  *dst = *src;
}

// ---------------- gather shift rows / zero unflagged ----------------
__global__ void shift_kernel(const float* __restrict__ x,
                             const int* __restrict__ mask,
                             const unsigned long long* __restrict__ keys,
                             float* __restrict__ out) {
  int i = blockIdx.x * blockDim.x + threadIdx.x;   // b*HW + p
  if (i >= BZ * HW) return;
  int b = i >> 12, p = i & (HW - 1);
  float* dst = out + (size_t)b * (CH + C2) * HW + (size_t)CH * HW + p;
  if (mask[p] != 1) {
    for (int c = 0; c < C2; ++c) dst[(size_t)c * HW] = 0.0f;
    return;
  }
  unsigned long long k = keys[i];
  unsigned int q = 0xFFFFFFFFu - (unsigned int)(k & 0xFFFFFFFFULL);
  if (q >= HW) q = 0;   // never-updated fallback (no valid q anywhere)
  const float* src = x + (size_t)b * CH * HW + (size_t)C2 * HW + q;
  for (int c = 0; c < C2; ++c) dst[(size_t)c * HW] = src[(size_t)c * HW];
}

extern "C" void kernel_launch(void* const* d_in, const int* in_sizes, int n_in,
                              void* d_out, int out_size, void* d_ws, size_t ws_size,
                              hipStream_t stream) {
  const float* x    = (const float*)d_in[0];
  const int*   mask = (const int*)d_in[1];
  float*       out  = (float*)d_out;

  char* ws = (char*)d_ws;
  float*              invn     = (float*)ws;                           // 128 KiB
  unsigned long long* keys     = (unsigned long long*)(ws + 131072);   // 256 KiB
  int*                pidx     = (int*)(ws + 131072 + 262144);         // 16 KiB
  int*                qbase    = (int*)(ws + 131072 + 262144 + 16384); // 1 KiB
  int*                counters = (int*)(ws + 131072 + 262144 + 16384 + 1024);

  init_ws_kernel<<<(BZ * HW + 255) / 256, 256, 0, stream>>>(keys, counters);
  compact_kernel<<<(HW + 255) / 256, 256, 0, stream>>>(mask, pidx, qbase, counters);
  norm_kernel<<<(BZ * HW + 255) / 256, 256, 0, stream>>>(x, invn);

  dim3 grid(BZ * 32, QS);
  gemm_argmax_kernel<<<grid, NT, 0, stream>>>(x, mask, invn, pidx, qbase,
                                              counters, keys);

  copy_kernel<<<(BZ * CH * HW / 4 + 255) / 256, 256, 0, stream>>>(x, out);
  shift_kernel<<<(BZ * HW + 255) / 256, 256, 0, stream>>>(x, mask, keys, out);
}
